// AttentionWithBias_37417755083152
// MI455X (gfx1250) — matrix operand order repaired and checked by reading, not compiled, not run on hardware
//
#include <hip/hip_runtime.h>
#include <hip/hip_fp16.h>

// AttentionWithBias for MI455X (gfx1250, wave32, WMMA).
// Bandwidth-bound: bias tensor (302MB fp32) read once; logits written once
// (QK^T fused into bias pass via LDS + WMMA C-operand); softmax fused into
// attn@V with online (flash) normalization so probs never touch HBM.

#define L 768
#define D_IN 256
#define D_BIAS 128
#define NH 8
#define DH 32

typedef __attribute__((ext_vector_type(16))) _Float16 v16h;
typedef __attribute__((ext_vector_type(8)))  _Float16 h8;
typedef __attribute__((ext_vector_type(8)))  float    v8f;
typedef __attribute__((ext_vector_type(4)))  float    v4f;

__device__ __forceinline__ v16h cat_h8(h8 lo, h8 hi) {
    v16h a;
#pragma unroll
    for (int i = 0; i < 8; ++i) { a[i] = lo[i]; a[i + 8] = hi[i]; }
    return a;
}

// A-tile (16 x K=32, f16, row-major source): lane l -> row (l&15); K halves
// [k0+kb..+7] and [k0+16+kb..+7], kb=(l>>4)*8  (ISA 7.12.2 16-bit A layout)
__device__ __forceinline__ v16h load_a16(const _Float16* __restrict__ p, int stride,
                                         int row0, int k0, int lane) {
    int m = lane & 15;
    int kb = (lane >> 4) << 3;
    const _Float16* r = p + (size_t)(row0 + m) * stride + k0 + kb;
    h8 lo = *(const h8*)(r);
    h8 hi = *(const h8*)(r + 16);
    return cat_h8(lo, hi);
}

// B-tile (K=32 x 16): lane l -> column (l&15); K halves [k0+kb..+7] and
// [k0+16+kb..+7], kb=(l>>4)*8, from source laid out [col][k]
__device__ __forceinline__ v16h load_b16(const _Float16* __restrict__ p, int stride,
                                         int col0, int k0, int lane) {
    int n = lane & 15;
    int kh = (lane >> 4) << 3;
    const _Float16* r = p + (size_t)(col0 + n) * stride + k0 + kh;
    h8 lo = *(const h8*)(r);
    h8 hi = *(const h8*)(r + 16);
    return cat_h8(lo, hi);
}

__device__ __forceinline__ v8f wmma16(v16h a, v16h b, v8f c) {
    return __builtin_amdgcn_wmma_f32_16x16x32_f16(false, a, false, b, (short)0, c, false, false);
}

// ---------- weight conversion: W[k][n] f32 -> WT[n][k] f16 (scale folded) ----------
__global__ void k_cvt_w(const float* __restrict__ W, _Float16* __restrict__ WT, float scale) {
    int k = blockIdx.x;
    int n = threadIdx.x;
    WT[(size_t)n * 256 + k] = (_Float16)(W[(size_t)k * 256 + n] * scale);
}

// Wb (128x8) -> WbT (16x128) f16, cols 8..15 zero-padded
__global__ void k_cvt_wb(const float* __restrict__ Wb, _Float16* __restrict__ WbT) {
    int col = blockIdx.x;   // 0..15
    int k = threadIdx.x;    // 0..127
    WbT[col * 128 + k] = (col < 8) ? (_Float16)Wb[k * 8 + col] : (_Float16)0.0f;
}

// ---------- layernorm of x -> xn f16 row-major [row][256] ----------
__global__ void k_ln_x(const float* __restrict__ x, const float* __restrict__ w,
                       const float* __restrict__ b, _Float16* __restrict__ xn) {
    int row = blockIdx.x;
    int c = threadIdx.x;            // 256 threads
    int wid = c >> 5;
    float v = x[(size_t)row * 256 + c];
    __shared__ float red[8];
    float s = v;
#pragma unroll
    for (int o = 16; o > 0; o >>= 1) s += __shfl_xor(s, o, 32);
    if ((c & 31) == 0) red[wid] = s;
    __syncthreads();
    float tot = 0.f;
#pragma unroll
    for (int i = 0; i < 8; ++i) tot += red[i];
    float mean = tot * (1.0f / 256.0f);
    __syncthreads();
    float d = v - mean;
    s = d * d;
#pragma unroll
    for (int o = 16; o > 0; o >>= 1) s += __shfl_xor(s, o, 32);
    if ((c & 31) == 0) red[wid] = s;
    __syncthreads();
    tot = 0.f;
#pragma unroll
    for (int i = 0; i < 8; ++i) tot += red[i];
    float var = tot * (1.0f / 256.0f);
    float rstd = rsqrtf(var + 1e-5f);
    xn[(size_t)row * 256 + c] = (_Float16)(d * rstd * w[c] + b[c]);
}

// ---------- q,k,v,gate projections (xn @ WT) : one 16x16 tile per wave ----------
__global__ void k_proj(const _Float16* __restrict__ xn,
                       const _Float16* __restrict__ WqT, const _Float16* __restrict__ WkT,
                       const _Float16* __restrict__ WvT, const _Float16* __restrict__ WgT,
                       const float* __restrict__ bg,
                       _Float16* __restrict__ qh, _Float16* __restrict__ kh,
                       _Float16* __restrict__ vT, float* __restrict__ gate) {
    int wid = threadIdx.x >> 5;
    int lane = threadIdx.x & 31;
    int mt = blockIdx.x * 4 + wid;      // 48 M-tiles
    int nt = blockIdx.y;                // 16 N-tiles
    int wsel = blockIdx.z;              // q,k,v,g
    const _Float16* WT = (wsel == 0) ? WqT : (wsel == 1) ? WkT : (wsel == 2) ? WvT : WgT;
    v8f c = {};
#pragma unroll
    for (int kt = 0; kt < 8; ++kt) {
        v16h a = load_a16(xn, 256, mt * 16, kt * 32, lane);
        v16h b = load_b16(WT, 256, nt * 16, kt * 32, lane);
        c = wmma16(a, b, c);
    }
    int n = lane & 15, hs = lane >> 4;
    int col = nt * 16 + n;
    if (wsel == 2) {                    // v stored transposed [col][row]
        h8 o;
#pragma unroll
        for (int j = 0; j < 8; ++j) o[j] = (_Float16)c[j];
        *(h8*)(vT + (size_t)col * 768 + mt * 16 + hs * 8) = o;
        return;
    }
    __shared__ float t[4][16][17];
#pragma unroll
    for (int j = 0; j < 8; ++j) t[wid][j + hs * 8][n] = c[j];
    __syncthreads();
    int m = n, cb = hs * 8;
    if (wsel == 3) {                    // gate = sigmoid(out + bg), f32 row-major
        v4f lo, hi;
#pragma unroll
        for (int j = 0; j < 4; ++j) {
            float g0 = t[wid][m][cb + j] + bg[nt * 16 + cb + j];
            float g1 = t[wid][m][cb + 4 + j] + bg[nt * 16 + cb + 4 + j];
            lo[j] = 1.0f / (1.0f + __expf(-g0));
            hi[j] = 1.0f / (1.0f + __expf(-g1));
        }
        float* dst = gate + (size_t)(mt * 16 + m) * 256 + nt * 16 + cb;
        *(v4f*)dst = lo;
        *(v4f*)(dst + 4) = hi;
    } else {                            // q/k row-major f16
        h8 o;
#pragma unroll
        for (int j = 0; j < 8; ++j) o[j] = (_Float16)t[wid][m][cb + j];
        _Float16* dst = ((wsel == 0) ? qh : kh) + (size_t)(mt * 16 + m) * 256 + nt * 16 + cb;
        *(h8*)dst = o;
    }
}

// ---------- fused: attn[h][q][k] = LN(bias[q,k,:]) @ Wb  +  K.Q^T ----------
// Block = 8 waves, covers (16 q) x (16 k). Phase 1: each wave LN+projects
// 2x16 bias rows (4 WMMAs each) into LDS logits [h][q][k]. Phase 2: wave h
// runs K.Q^T for head h with the LDS logits as WMMA C operand -> attn written
// exactly once, contiguous b128 stores.
__global__ void k_bias_qk(const float* __restrict__ bias, const float* __restrict__ lw,
                          const float* __restrict__ lb, const _Float16* __restrict__ WbT,
                          const _Float16* __restrict__ qh, const _Float16* __restrict__ kh,
                          float* __restrict__ attn) {
    int wid = threadIdx.x >> 5;
    int lane = threadIdx.x & 31;
    int qt = blockIdx.x;                // 48
    int kt = blockIdx.y;                // 48
    int m = lane & 15, hs = lane >> 4;
    int kb = hs * 8;
    __shared__ float lg[8][16][17];     // [head][q in tile][k in tile]

    // prefetch this wave's second task's row stream
    __builtin_prefetch(bias + ((size_t)(qt * 16 + wid + 8) * 768 + kt * 16 + m) * 128, 0, 0);

#pragma unroll
    for (int r = 0; r < 2; ++r) {
        int ql = wid + 8 * r;           // q within tile, 0..15
        const float* row = bias + ((size_t)(qt * 16 + ql) * 768 + kt * 16 + m) * 128;
        v4f xd[16];
#pragma unroll
        for (int s = 0; s < 4; ++s) {
            const float* p0 = row + s * 32 + kb;
            const float* p1 = row + s * 32 + 16 + kb;
            xd[s * 4 + 0] = *(const v4f*)(p0);
            xd[s * 4 + 1] = *(const v4f*)(p0 + 4);
            xd[s * 4 + 2] = *(const v4f*)(p1);
            xd[s * 4 + 3] = *(const v4f*)(p1 + 4);
        }
        float sum = 0.f, ss = 0.f;
#pragma unroll
        for (int i = 0; i < 16; ++i)
#pragma unroll
            for (int j = 0; j < 4; ++j) { float u = xd[i][j]; sum += u; ss += u * u; }
        sum += __shfl_xor(sum, 16, 32);
        ss += __shfl_xor(ss, 16, 32);
        float mean = sum * (1.0f / 128.0f);
        float var = ss * (1.0f / 128.0f) - mean * mean;
        float rstd = rsqrtf(var + 1e-5f);
        v8f c = {};
#pragma unroll
        for (int s = 0; s < 4; ++s) {
            int c0 = s * 32 + kb, c1 = s * 32 + 16 + kb;
            v4f w0 = *(const v4f*)(lw + c0), w1 = *(const v4f*)(lw + c0 + 4);
            v4f w2 = *(const v4f*)(lw + c1), w3 = *(const v4f*)(lw + c1 + 4);
            v4f b0 = *(const v4f*)(lb + c0), b1 = *(const v4f*)(lb + c0 + 4);
            v4f b2 = *(const v4f*)(lb + c1), b3 = *(const v4f*)(lb + c1 + 4);
            v16h a;
#pragma unroll
            for (int j = 0; j < 4; ++j) {
                a[j]      = (_Float16)((xd[s * 4 + 0][j] - mean) * rstd * w0[j] + b0[j]);
                a[4 + j]  = (_Float16)((xd[s * 4 + 1][j] - mean) * rstd * w1[j] + b1[j]);
                a[8 + j]  = (_Float16)((xd[s * 4 + 2][j] - mean) * rstd * w2[j] + b2[j]);
                a[12 + j] = (_Float16)((xd[s * 4 + 3][j] - mean) * rstd * w3[j] + b3[j]);
            }
            v16h bm = load_b16(WbT, 128, 0, s * 32, lane);
            c = wmma16(a, bm, c);
        }
        // D: (M = k offset j+8*hs, N = head n); heads 0..7 valid
        int n = lane & 15;
        if (n < 8) {
#pragma unroll
            for (int j = 0; j < 8; ++j) lg[n][ql][8 * hs + j] = c[j];
        }
    }
    __syncthreads();

    // phase 2: wave wid = head; D = K.Q^T + bias-logits (C from LDS)
    int h = wid;
    int n = lane & 15;
    v8f c;
#pragma unroll
    for (int j = 0; j < 8; ++j) c[j] = lg[h][n][8 * hs + j];
    v16h a = load_a16(kh, 256, kt * 16, h * 32, lane);   // M = k rows
    v16h b = load_b16(qh, 256, qt * 16, h * 32, lane);   // N = q rows
    c = wmma16(a, b, c);
    float* dst = attn + ((size_t)h * 768 + qt * 16 + n) * 768 + kt * 16 + hs * 8;
    v4f lo = {c[0], c[1], c[2], c[3]};
    v4f hi = {c[4], c[5], c[6], c[7]};
    *(v4f*)dst = lo;
    *(v4f*)(dst + 4) = hi;
}

// ---------- fused online softmax + attn@V + gate: gout[q][h*32+dh] f16 ----------
// One wave per (q-tile, head). Streams f32 logits in A-fragment order (lane
// pair l/l^16 covers a full 32-k chunk of one q row), keeps running (m,l),
// rescales both D accumulators, builds f16 prob A-fragment in-register.
__global__ void k_av_soft(const float* __restrict__ attn, const _Float16* __restrict__ vT,
                          const float* __restrict__ gate, _Float16* __restrict__ gout) {
    int wid = threadIdx.x >> 5;
    int lane = threadIdx.x & 31;
    int qt = blockIdx.x * 4 + wid;      // 48 q-tiles
    int h = blockIdx.y;                 // 8 heads
    int ln15 = lane & 15, hs = lane >> 4;
    int kb = hs * 8;
    const float* arow = attn + ((size_t)h * 768 + qt * 16 + ln15) * 768;
    v8f acc0 = {}, acc1 = {};
    float mrun = -1e30f, lrun = 0.f;
    for (int kt = 0; kt < 24; ++kt) {   // K = 768 keys
        const float* p0 = arow + kt * 32 + kb;
        const float* p1 = arow + kt * 32 + 16 + kb;
        v4f x0 = *(const v4f*)p0, x1 = *(const v4f*)(p0 + 4);
        v4f x2 = *(const v4f*)p1, x3 = *(const v4f*)(p1 + 4);
        float xs[16];
#pragma unroll
        for (int j = 0; j < 4; ++j) {
            xs[j] = x0[j]; xs[4 + j] = x1[j]; xs[8 + j] = x2[j]; xs[12 + j] = x3[j];
        }
        float tmax = xs[0];
#pragma unroll
        for (int i = 1; i < 16; ++i) tmax = fmaxf(tmax, xs[i]);
        tmax = fmaxf(tmax, __shfl_xor(tmax, 16, 32));
        float mnew = fmaxf(mrun, tmax);
        float corr = __expf(mrun - mnew);
        v16h a;
        float tsum = 0.f;
#pragma unroll
        for (int i = 0; i < 16; ++i) {
            float p = __expf(xs[i] - mnew);
            tsum += p;
            a[i] = (_Float16)p;
        }
        tsum += __shfl_xor(tsum, 16, 32);
        lrun = lrun * corr + tsum;
        mrun = mnew;
#pragma unroll
        for (int j = 0; j < 8; ++j) {   // rescale accumulators (per q row j+8*hs)
            float cj = __shfl(corr, 8 * hs + j, 32);
            acc0[j] *= cj;
            acc1[j] *= cj;
        }
        v16h b0 = load_b16(vT, 768, h * 32, kt * 32, lane);
        v16h b1 = load_b16(vT, 768, h * 32 + 16, kt * 32, lane);
        acc0 = wmma16(a, b0, acc0);
        acc1 = wmma16(a, b1, acc1);
    }
    // epilogue: /l, *gate, transpose to row-major f16
    __shared__ float t[4][16][17];
#pragma unroll
    for (int r = 0; r < 2; ++r) {
        v8f acc = r ? acc1 : acc0;
        int col = h * 32 + r * 16 + ln15;       // dh column owned by this lane
#pragma unroll
        for (int j = 0; j < 8; ++j) {
            int qrow = qt * 16 + 8 * hs + j;
            float lj = __shfl(lrun, 8 * hs + j, 32);
            float o = (acc[j] / lj) * gate[(size_t)qrow * 256 + col];
            t[wid][8 * hs + j][ln15] = o;
        }
        __syncthreads();
        h8 oo;
#pragma unroll
        for (int j = 0; j < 8; ++j) oo[j] = (_Float16)t[wid][ln15][kb + j];
        *(h8*)(gout + (size_t)(qt * 16 + ln15) * 256 + h * 32 + r * 16 + kb) = oo;
        __syncthreads();
    }
}

// ---------- final projection: out = gout @ Wo + bo (f32) ----------
__global__ void k_out(const _Float16* __restrict__ gout, const _Float16* __restrict__ WoT,
                      const float* __restrict__ bo, float* __restrict__ out) {
    int wid = threadIdx.x >> 5;
    int lane = threadIdx.x & 31;
    int mt = blockIdx.x * 4 + wid;      // 48
    int nt = blockIdx.y;                // 16
    v8f c = {};
#pragma unroll
    for (int kt = 0; kt < 8; ++kt) {
        v16h a = load_a16(gout, 256, mt * 16, kt * 32, lane);
        v16h b = load_b16(WoT, 256, nt * 16, kt * 32, lane);
        c = wmma16(a, b, c);
    }
    int n = lane & 15, hs = lane >> 4;
    __shared__ float t[4][16][17];
#pragma unroll
    for (int j = 0; j < 8; ++j) t[wid][j + hs * 8][n] = c[j];
    __syncthreads();
    int m = n, cb = hs * 8;
    v4f lo, hi;
#pragma unroll
    for (int j = 0; j < 4; ++j) {
        lo[j] = t[wid][m][cb + j] + bo[nt * 16 + cb + j];
        hi[j] = t[wid][m][cb + 4 + j] + bo[nt * 16 + cb + 4 + j];
    }
    float* dst = out + (size_t)(mt * 16 + m) * 256 + nt * 16 + cb;
    *(v4f*)dst = lo;
    *(v4f*)(dst + 4) = hi;
}

extern "C" void kernel_launch(void* const* d_in, const int* in_sizes, int n_in,
                              void* d_out, int out_size, void* d_ws, size_t ws_size,
                              hipStream_t stream) {
    (void)in_sizes; (void)n_in; (void)out_size; (void)ws_size;
    const float* X    = (const float*)d_in[0];
    const float* BIAS = (const float*)d_in[1];
    const float* LNxW = (const float*)d_in[2];
    const float* LNxB = (const float*)d_in[3];
    const float* LNbW = (const float*)d_in[4];
    const float* LNbB = (const float*)d_in[5];
    const float* Wq   = (const float*)d_in[6];
    const float* Wk   = (const float*)d_in[7];
    const float* Wv   = (const float*)d_in[8];
    const float* Wb   = (const float*)d_in[9];
    const float* Wg   = (const float*)d_in[10];
    const float* bg   = (const float*)d_in[11];
    const float* Wo   = (const float*)d_in[12];
    const float* bo   = (const float*)d_in[13];
    float* OUT = (float*)d_out;

    char* ws = (char*)d_ws;
    size_t off = 0;
    auto alloc = [&](size_t bytes) {
        void* p = ws + off;
        off += (bytes + 255) & ~(size_t)255;
        return p;
    };
    _Float16* xn   = (_Float16*)alloc(768 * 256 * 2);
    _Float16* WqT  = (_Float16*)alloc(256 * 256 * 2);
    _Float16* WkT  = (_Float16*)alloc(256 * 256 * 2);
    _Float16* WvT  = (_Float16*)alloc(256 * 256 * 2);
    _Float16* WgT  = (_Float16*)alloc(256 * 256 * 2);
    _Float16* WoT  = (_Float16*)alloc(256 * 256 * 2);
    _Float16* WbT  = (_Float16*)alloc(16 * 128 * 2);
    _Float16* qh   = (_Float16*)alloc(768 * 256 * 2);
    _Float16* kh   = (_Float16*)alloc(768 * 256 * 2);
    _Float16* vT   = (_Float16*)alloc(256 * 768 * 2);
    float*    gate = (float*)   alloc(768 * 256 * 4);
    _Float16* gout = (_Float16*)alloc(768 * 256 * 2);
    float*    attn = (float*)   alloc((size_t)8 * 768 * 768 * 4);

    const float kscale = 0.17677669529663687f;  // 1/sqrt(DH)

    k_cvt_w<<<dim3(256), dim3(256), 0, stream>>>(Wq, WqT, 1.0f);
    k_cvt_w<<<dim3(256), dim3(256), 0, stream>>>(Wk, WkT, kscale);
    k_cvt_w<<<dim3(256), dim3(256), 0, stream>>>(Wv, WvT, 1.0f);
    k_cvt_w<<<dim3(256), dim3(256), 0, stream>>>(Wg, WgT, 1.0f);
    k_cvt_w<<<dim3(256), dim3(256), 0, stream>>>(Wo, WoT, 1.0f);
    k_cvt_wb<<<dim3(16), dim3(128), 0, stream>>>(Wb, WbT);

    k_ln_x<<<dim3(768), dim3(256), 0, stream>>>(X, LNxW, LNxB, xn);

    k_proj<<<dim3(12, 16, 4), dim3(128), 0, stream>>>(xn, WqT, WkT, WvT, WgT, bg,
                                                      qh, kh, vT, gate);

    k_bias_qk<<<dim3(48, 48), dim3(256), 0, stream>>>(BIAS, LNbW, LNbB, WbT, qh, kh, attn);

    k_av_soft<<<dim3(12, 8), dim3(128), 0, stream>>>(attn, vT, gate, gout);

    k_out<<<dim3(12, 16), dim3(128), 0, stream>>>(gout, WoT, bo, OUT);
}
